// DeltaNet_31877247271572
// MI455X (gfx1250) — compile-verified, matrix operand order checked by reading, not run on hardware
//
#include <hip/hip_runtime.h>
#include <hip/hip_bf16.h>
#include <math.h>

typedef __attribute__((ext_vector_type(2)))  float  v2f;
typedef __attribute__((ext_vector_type(8)))  float  v8f;
typedef __attribute__((ext_vector_type(8)))  __bf16 v8bf;
typedef __attribute__((ext_vector_type(16))) __bf16 v16bf;

#define DEV __device__ __forceinline__

// ---------------- fragment helpers ----------------

// bf16 A/B fragment for V_WMMA_F32_16X16X32_BF16.
// Operand matrix stored row-major with leading dim `ld`; for the B operand we
// pre-transpose weights to (N,K) so "row" = output column, giving identical
// contiguous loads. Lane l (l<16) holds row m=l, K {k0..k0+7, k0+16..k0+23};
// lane l+16 holds K {k0+8..k0+15, k0+24..k0+31}.
DEV v16bf ldfrag_bf16(const __bf16* base, int ld, int row, int k0, int lane) {
  int half = lane >> 4;
  const __bf16* p = base + (size_t)row * ld + k0 + half * 8;
  v8bf lo = *(const v8bf*)(p);
  v8bf hi = *(const v8bf*)(p + 16);
  v16bf r;
#pragma unroll
  for (int i = 0; i < 8; ++i) { r[i] = lo[i]; r[i + 8] = hi[i]; }
  return r;
}

// f32 A fragment (16x4): lane holds row m=(l&15), K = k0+2*half {+0,+1}
DEV v2f afrag_f32(const float* M, int ld, int row, int k0, int lane) {
  int half = lane >> 4;
  const float* p = M + row * ld + k0 + 2 * half;
  v2f a; a[0] = p[0]; a[1] = p[1];
  return a;
}
// f32 B fragment (4x16): lane holds col n=(l&15), K = k0+2*half {+0,+1}
DEV v2f bfrag_f32(const float* M, int ld, int col, int k0, int lane) {
  int half = lane >> 4;
  const float* p = M + (k0 + 2 * half) * ld + col;
  v2f b; b[0] = p[0]; b[1] = p[ld];
  return b;
}

DEV float block_sum_256(float v, float* red) {
  int tid = threadIdx.x;
  red[tid] = v;
  __syncthreads();
  for (int s = 128; s > 0; s >>= 1) {
    if (tid < s) red[tid] += red[tid + s];
    __syncthreads();
  }
  float r = red[0];
  __syncthreads();
  return r;
}

// ---------------- conversions ----------------

__global__ void convert_hs_kernel(const float* __restrict__ hs, __bf16* __restrict__ hsb,
                                  __bf16* __restrict__ gate_in, int BL, int D) {
  size_t i = (size_t)blockIdx.x * blockDim.x + threadIdx.x;
  float x = hs[i];
  hsb[i] = (__bf16)x;
  size_t r = i / D, c = i % D;
  gate_in[r * 1120 + c] = (__bf16)x;
}

// W (K,N) f32 -> Wt (N,K) bf16
__global__ void convert_wT_kernel(const float* __restrict__ W, __bf16* __restrict__ Wt,
                                  int K, int N) {
  size_t i = (size_t)blockIdx.x * blockDim.x + threadIdx.x;
  size_t kk = i / N, n = i % N;
  Wt[n * (size_t)K + kk] = (__bf16)W[i];
}

// ---------------- bf16 WMMA GEMM: C(M,N)=A(M,K)@B(K,N), Bt stored (N,K) -----
// 128x128 block, 8 waves (4x2), each wave a 32x64 tile (2x4 WMMA frags).
// EPI=0: store f32 to C.  EPI=1: x+bias -> exact GELU -> store bf16 to Cb.

template <int EPI>
__global__ void gemm_bf16_kernel(const __bf16* __restrict__ A, const __bf16* __restrict__ Bt,
                                 float* __restrict__ C, __bf16* __restrict__ Cb,
                                 const float* __restrict__ bias, int M, int N, int K) {
  int lane = threadIdx.x & 31, wave = threadIdx.x >> 5;
  int wm = wave & 3, wn = wave >> 2;  // 4x2 waves
  int bm = blockIdx.y * 128, bn = blockIdx.x * 128;
  int rm = bm + wm * 32;   // rows rm..rm+31
  int rn = bn + wn * 64;   // cols rn..rn+63
  const v8f vzero = {};
  v8f acc[2][4];
#pragma unroll
  for (int i = 0; i < 2; ++i)
#pragma unroll
    for (int j = 0; j < 4; ++j) acc[i][j] = vzero;

  for (int k0 = 0; k0 < K; k0 += 32) {
    if (k0 + 64 < K) {  // speculative prefetch of next-next k-tile
      __builtin_prefetch(A + (size_t)(rm + (lane & 15)) * K + k0 + 64, 0, 1);
      __builtin_prefetch(Bt + (size_t)(rn + (lane & 15)) * K + k0 + 64, 0, 1);
    }
    v16bf a[2], b[4];
#pragma unroll
    for (int i = 0; i < 2; ++i)
      a[i] = ldfrag_bf16(A, K, rm + i * 16 + (lane & 15), k0, lane);
#pragma unroll
    for (int j = 0; j < 4; ++j)
      b[j] = ldfrag_bf16(Bt, K, rn + j * 16 + (lane & 15), k0, lane);
#pragma unroll
    for (int i = 0; i < 2; ++i)
#pragma unroll
      for (int j = 0; j < 4; ++j)
        acc[i][j] = __builtin_amdgcn_wmma_f32_16x16x32_bf16(false, a[i], false, b[j],
                                                            (short)0, acc[i][j], false, false);
  }

  int half = lane >> 4, n = lane & 15;
#pragma unroll
  for (int i = 0; i < 2; ++i) {
#pragma unroll
    for (int j = 0; j < 4; ++j) {
#pragma unroll
      for (int r = 0; r < 8; ++r) {
        int row = rm + i * 16 + r + 8 * half;
        int col = rn + j * 16 + n;
        if (EPI == 0) {
          C[(size_t)row * N + col] = acc[i][j][r];
        } else {
          float x = acc[i][j][r] + bias[col];
          float g = 0.5f * x * (1.f + erff(x * 0.70710678118654752f));
          Cb[(size_t)row * N + col] = (__bf16)g;
        }
      }
    }
  }
}

// ---------------- causal depthwise conv (k=4) + SiLU ----------------

__global__ void conv_silu_kernel(const float* __restrict__ lin, const float* __restrict__ cw,
                                 float* __restrict__ out, int B, int L, int D) {
  size_t i = (size_t)blockIdx.x * blockDim.x + threadIdx.x;
  int c = (int)(i % D);
  size_t bl = i / D;
  int l = (int)(bl % L), b = (int)(bl / L);
  float acc = 0.f;
#pragma unroll
  for (int t = 0; t < 4; ++t) {
    int li = l - 3 + t;
    if (li >= 0) acc += cw[c * 4 + t] * lin[((size_t)(b * L + li)) * D + c];
  }
  out[i] = acc / (1.f + expf(-acc));  // silu
}

// ---------------- beta = sigmoid(hs @ b_w) ----------------

__global__ void beta_kernel(const float* __restrict__ hs, const float* __restrict__ bw,
                            float* __restrict__ beta, int BL, int D, int H) {
  int i = blockIdx.x * blockDim.x + threadIdx.x;  // BL*H
  int h = i % H, bl = i / H;
  const float* row = hs + (size_t)bl * D;
  float acc = 0.f;
  for (int c = 0; c < D; ++c) acc += row[c] * bw[c * H + h];
  beta[i] = 1.f / (1.f + expf(-acc));
}

// ---------------- delta rule, chunkwise, dv sliced by 64 ----------------
// grid: (B*H, 4 slices), 256 threads (8 waves). S slice kept f32 in LDS.

__global__ void delta_kernel(const float* __restrict__ qg, const float* __restrict__ kg,
                             const float* __restrict__ vg, const float* __restrict__ betag,
                             float* __restrict__ outg, int B, int L, int H) {
  extern __shared__ float lds[];
  float* S   = lds;              // 256*64
  float* qs  = S   + 256 * 64;   // 32*256
  float* ks  = qs  + 32 * 256;   // 32*256
  float* kbs = ks  + 32 * 256;   // 32*256
  float* wms = kbs + 32 * 256;   // 32*256
  float* us  = wms + 32 * 256;   // 32*64
  float* uhs = us  + 32 * 64;    // 32*64  (also holds beta*v before u_hat)
  float* Tm  = uhs + 32 * 64;    // 32*32
  float* Am  = Tm  + 32 * 32;    // 32*32  (kb @ k^T)
  float* Qk  = Am  + 32 * 32;    // 32*32  (q @ k^T)
  float* bv  = Qk  + 32 * 32;    // 32

  const int tid = threadIdx.x, lane = tid & 31, wave = tid >> 5;
  const int b = blockIdx.x >> 2, h = blockIdx.x & 3, sl = blockIdx.y;
  const int baseCol = h * 256;
  const int half = lane >> 4, ln = lane & 15;

  for (int i = tid; i < 256 * 64; i += 256) S[i] = 0.f;
  __syncthreads();

  const int nchunks = L / 32;
  for (int c = 0; c < nchunks; ++c) {
    const int l0 = c * 32;
    // load q,k chunk + beta
    for (int i = tid; i < 32 * 256; i += 256) {
      int r = i >> 8, cc = i & 255;
      size_t g = ((size_t)(b * L + l0 + r)) * 1024 + baseCol + cc;
      qs[i] = qg[g];
      ks[i] = kg[g];
    }
    if (tid < 32) bv[tid] = betag[(size_t)(b * L + l0 + tid) * H + h];
    __syncthreads();
    // l2norm rows of q (threads 0-31) and k (threads 32-63)
    if (tid < 64) {
      int r = tid & 31;
      float* M = (tid < 32) ? qs : ks;
      float s = 0.f;
      for (int cc = 0; cc < 256; ++cc) { float x = M[r * 256 + cc]; s += x * x; }
      float inv = rsqrtf(s + 1e-6f);
      for (int cc = 0; cc < 256; ++cc) M[r * 256 + cc] *= inv;
    }
    __syncthreads();
    // kb = k*beta ; v*beta -> uhs (temp)
    for (int i = tid; i < 32 * 256; i += 256) kbs[i] = ks[i] * bv[i >> 8];
    for (int i = tid; i < 32 * 64; i += 256) {
      int r = i >> 6, cc = i & 63;
      uhs[i] = vg[((size_t)(b * L + l0 + r)) * 1024 + baseCol + sl * 64 + cc] * bv[r];
    }
    __syncthreads();
    // kb@k^T -> Am (waves 0-3), q@k^T -> Qk (waves 4-7);  B = k^T -> read rows of k
    {
      int t = wave & 3, mt = t >> 1, nt = t & 1;
      const float* Aop = (wave < 4) ? kbs : qs;
      float* Cop = (wave < 4) ? Am : Qk;
      v8f acc = {};
      for (int k0 = 0; k0 < 256; k0 += 4) {
        v2f a = afrag_f32(Aop, 256, mt * 16 + ln, k0, lane);
        v2f bb = afrag_f32(ks, 256, nt * 16 + ln, k0, lane);
        acc = __builtin_amdgcn_wmma_f32_16x16x4_f32(false, a, false, bb, (short)0, acc, false, false);
      }
#pragma unroll
      for (int r = 0; r < 8; ++r)
        Cop[(mt * 16 + r + 8 * half) * 32 + nt * 16 + ln] = acc[r];
    }
    __syncthreads();
    // T = (I - A)^{-1}, A = strict_lower(-Am): forward substitution
    for (int i = 0; i < 32; ++i) {
      if (tid < 32) {
        float acc = (i == tid) ? 1.f : 0.f;
        for (int j = 0; j < i; ++j) acc += (-Am[i * 32 + j]) * Tm[j * 32 + tid];
        Tm[i * 32 + tid] = acc;
      }
      __syncthreads();
    }
    // u = T @ (beta*v) -> us   (2x4 tiles, one per wave)
    {
      int mt = wave & 1, nt = wave >> 1;
      v8f acc = {};
      for (int k0 = 0; k0 < 32; k0 += 4) {
        v2f a = afrag_f32(Tm, 32, mt * 16 + ln, k0, lane);
        v2f bb = bfrag_f32(uhs, 64, nt * 16 + ln, k0, lane);
        acc = __builtin_amdgcn_wmma_f32_16x16x4_f32(false, a, false, bb, (short)0, acc, false, false);
      }
#pragma unroll
      for (int r = 0; r < 8; ++r)
        us[(mt * 16 + r + 8 * half) * 64 + nt * 16 + ln] = acc[r];
      // w = T @ kb -> wms (2x16 tiles, 4 per wave)
      for (int it = 0; it < 4; ++it) {
        int t = wave + 8 * it;
        int wmt = t & 1, wnt = t >> 1;
        v8f a2 = {};
        for (int k0 = 0; k0 < 32; k0 += 4) {
          v2f a = afrag_f32(Tm, 32, wmt * 16 + ln, k0, lane);
          v2f bb = bfrag_f32(kbs, 256, wnt * 16 + ln, k0, lane);
          a2 = __builtin_amdgcn_wmma_f32_16x16x4_f32(false, a, false, bb, (short)0, a2, false, false);
        }
#pragma unroll
        for (int r = 0; r < 8; ++r)
          wms[(wmt * 16 + r + 8 * half) * 256 + wnt * 16 + ln] = a2[r];
      }
    }
    __syncthreads();
    // u_hat = u - w @ S  -> uhs
    {
      int mt = wave & 1, nt = wave >> 1;
      v8f acc = {};
      for (int k0 = 0; k0 < 256; k0 += 4) {
        v2f a = afrag_f32(wms, 256, mt * 16 + ln, k0, lane);
        v2f bb = bfrag_f32(S, 64, nt * 16 + ln, k0, lane);
        acc = __builtin_amdgcn_wmma_f32_16x16x4_f32(false, a, false, bb, (short)0, acc, false, false);
      }
#pragma unroll
      for (int r = 0; r < 8; ++r) {
        int idx = (mt * 16 + r + 8 * half) * 64 + nt * 16 + ln;
        uhs[idx] = us[idx] - acc[r];
      }
    }
    __syncthreads();
    // o = q @ S + tril(qk) @ u_hat  -> global
    {
      int mt = wave & 1, nt = wave >> 1;
      v8f acc = {};
      for (int k0 = 0; k0 < 256; k0 += 4) {
        v2f a = afrag_f32(qs, 256, mt * 16 + ln, k0, lane);
        v2f bb = bfrag_f32(S, 64, nt * 16 + ln, k0, lane);
        acc = __builtin_amdgcn_wmma_f32_16x16x4_f32(false, a, false, bb, (short)0, acc, false, false);
      }
      int m8 = mt * 16 + ln;
      for (int k0 = 0; k0 < 32; k0 += 4) {
        int kk = k0 + 2 * half;
        v2f a;
        a[0] = (kk     <= m8) ? Qk[m8 * 32 + kk]     : 0.f;
        a[1] = (kk + 1 <= m8) ? Qk[m8 * 32 + kk + 1] : 0.f;
        v2f bb = bfrag_f32(uhs, 64, nt * 16 + ln, k0, lane);
        acc = __builtin_amdgcn_wmma_f32_16x16x4_f32(false, a, false, bb, (short)0, acc, false, false);
      }
#pragma unroll
      for (int r = 0; r < 8; ++r) {
        int row = mt * 16 + r + 8 * half;
        outg[((size_t)(b * L + l0 + row)) * 1024 + baseCol + sl * 64 + nt * 16 + ln] = acc[r];
      }
    }
    __syncthreads();
    // S += k^T @ u_hat   (16x4 tiles, 8 per wave); A = k^T -> bfrag over k
    for (int it = 0; it < 8; ++it) {
      int t = wave * 8 + it;
      int mt = t >> 2, nt = t & 3;
      v8f acc = {};
      for (int k0 = 0; k0 < 32; k0 += 4) {
        v2f a = bfrag_f32(ks, 256, mt * 16 + ln, k0, lane);
        v2f bb = bfrag_f32(uhs, 64, nt * 16 + ln, k0, lane);
        acc = __builtin_amdgcn_wmma_f32_16x16x4_f32(false, a, false, bb, (short)0, acc, false, false);
      }
#pragma unroll
      for (int r = 0; r < 8; ++r)
        S[(mt * 16 + r + 8 * half) * 64 + nt * 16 + ln] += acc[r];
    }
    __syncthreads();
  }
}

// ---------------- FIR branches + per-head stats -> gate_in tail ----------------

__global__ void fir_stats_kernel(const float* __restrict__ v, const float* __restrict__ dlt,
                                 const float* __restrict__ fw1, const float* __restrict__ fw3,
                                 const float* __restrict__ fw7, const float* __restrict__ fw31,
                                 float* __restrict__ f1o, float* __restrict__ f3o,
                                 float* __restrict__ f7o, float* __restrict__ f31o,
                                 __bf16* __restrict__ gate_in, int B, int L, int H) {
  __shared__ float red[256];
  int tid = threadIdx.x;
  int bl = blockIdx.x / H, h = blockIdx.x % H;
  int b = bl / L, l = bl % L;
  int c = h * 256 + tid;
  size_t rowBase = (size_t)bl * 1024 + c;

  auto fir = [&](const float* w, int kw) {
    float acc = 0.f;
    for (int t = 0; t < kw; ++t) {
      int li = l - (kw - 1) + t;
      if (li >= 0) acc += w[c * kw + t] * v[((size_t)(b * L + li)) * 1024 + c];
    }
    return acc;
  };
  float br[6];
  br[0] = fir(fw1, 1);
  br[1] = fir(fw3, 3);
  br[2] = fir(fw7, 7);
  br[3] = fir(fw31, 31);
  br[4] = dlt[rowBase];
  br[5] = v[rowBase];
  f1o[rowBase] = br[0]; f3o[rowBase] = br[1]; f7o[rowBase] = br[2]; f31o[rowBase] = br[3];

  for (int s = 0; s < 6; ++s) {
    float x = br[s];
    float s1 = block_sum_256(x, red);
    float s2 = block_sum_256(x * x, red);
    float sa = block_sum_256(fabsf(x), red);
    if (tid == 0) {
      float mean = s1 / 256.f;
      float var = (s2 - s1 * s1 / 256.f) / 255.f;
      float stdv = sqrtf(fmaxf(var, 0.f));
      size_t o = (size_t)bl * 1120 + 1024 + h * 24 + s * 4;
      gate_in[o + 0] = (__bf16)mean;
      gate_in[o + 1] = (__bf16)stdv;
      gate_in[o + 2] = (__bf16)(sa / 256.f);
      gate_in[o + 3] = (__bf16)sqrtf(s2);
    }
    __syncthreads();
  }
}

// ---------------- gate logits ----------------

__global__ void logits_kernel(const __bf16* __restrict__ hg, const float* __restrict__ w2,
                              const float* __restrict__ b2, float* __restrict__ logits,
                              int Kd, int Nd) {
  __shared__ float row[1024];
  int tid = threadIdx.x;
  size_t bl = blockIdx.x;
  for (int i = tid; i < Kd; i += 256) row[i] = (float)hg[bl * Kd + i];
  __syncthreads();
  if (tid < Nd) {
    float acc = b2[tid];
    for (int kk = 0; kk < Kd; ++kk) acc += row[kk] * w2[kk * Nd + tid];
    logits[bl * Nd + tid] = acc;
  }
}

// ---------------- softmax gate + fuse + RMSNorm ----------------

__global__ void fuse_kernel(const float* __restrict__ f1, const float* __restrict__ f3,
                            const float* __restrict__ f7, const float* __restrict__ f31,
                            const float* __restrict__ dlt, const float* __restrict__ v,
                            const float* __restrict__ logits, const float* __restrict__ glt,
                            const float* __restrict__ onw, __bf16* __restrict__ fused,
                            int B, int L, int H) {
  __shared__ float red[256];
  __shared__ float wsm[6];
  int tid = threadIdx.x;
  int bl = blockIdx.x / H, h = blockIdx.x % H;
  size_t rowBase = (size_t)bl * 1024 + h * 256 + tid;
  if (tid == 0) {
    float temp = log1pf(expf(glt[h])) + 0.5f;
    float lg[6], mx = -1e30f;
    for (int s = 0; s < 6; ++s) { lg[s] = logits[(size_t)bl * 24 + h * 6 + s] / temp; mx = fmaxf(mx, lg[s]); }
    float den = 0.f;
    for (int s = 0; s < 6; ++s) { lg[s] = expf(lg[s] - mx); den += lg[s]; }
    for (int s = 0; s < 6; ++s) wsm[s] = lg[s] / den;
  }
  __syncthreads();
  float x = wsm[0] * f1[rowBase] + wsm[1] * f3[rowBase] + wsm[2] * f7[rowBase] +
            wsm[3] * f31[rowBase] + wsm[4] * dlt[rowBase] + wsm[5] * v[rowBase];
  float s2 = block_sum_256(x * x, red);
  float inv = rsqrtf(s2 / 256.f + 1e-5f);
  fused[rowBase] = (__bf16)(x * inv * onw[tid]);
}

// ---------------- host launch ----------------

extern "C" void kernel_launch(void* const* d_in, const int* in_sizes, int n_in,
                              void* d_out, int out_size, void* d_ws, size_t ws_size,
                              hipStream_t stream) {
  (void)in_sizes; (void)n_in; (void)out_size; (void)ws_size;
  const int B = 2, L = 2048, D = 1024, H = 4;
  const int BL = B * L;  // 4096

  const float* hs   = (const float*)d_in[0];
  const float* qw   = (const float*)d_in[1];
  const float* kw   = (const float*)d_in[2];
  const float* vw   = (const float*)d_in[3];
  const float* bw   = (const float*)d_in[4];
  const float* qcw  = (const float*)d_in[5];
  const float* kcw  = (const float*)d_in[6];
  const float* vcw  = (const float*)d_in[7];
  const float* fw1  = (const float*)d_in[8];
  const float* fw3  = (const float*)d_in[9];
  const float* fw7  = (const float*)d_in[10];
  const float* fw31 = (const float*)d_in[11];
  const float* w1   = (const float*)d_in[12];
  const float* b1   = (const float*)d_in[13];
  const float* w2   = (const float*)d_in[14];
  const float* b2   = (const float*)d_in[15];
  const float* glt  = (const float*)d_in[16];
  const float* onw  = (const float*)d_in[17];
  const float* ow   = (const float*)d_in[18];
  float* out = (float*)d_out;

  char* ws = (char*)d_ws;
  size_t off = 0;
  auto alloc = [&](size_t bytes) -> char* {
    char* p = ws + off;
    off += (bytes + 255) & ~(size_t)255;
    return p;
  };

  __bf16* hs_bf   = (__bf16*)alloc((size_t)BL * D * 2);
  __bf16* gate_in = (__bf16*)alloc((size_t)BL * 1120 * 2);
  __bf16* wq_t    = (__bf16*)alloc((size_t)D * D * 2);
  __bf16* wk_t    = (__bf16*)alloc((size_t)D * D * 2);
  __bf16* wv_t    = (__bf16*)alloc((size_t)D * D * 2);
  __bf16* wo_t    = (__bf16*)alloc((size_t)D * D * 2);
  __bf16* w1_t    = (__bf16*)alloc((size_t)1120 * 1024 * 2);
  float*  q_lin   = (float*)alloc((size_t)BL * D * 4);
  float*  k_lin   = (float*)alloc((size_t)BL * D * 4);
  float*  v_lin   = (float*)alloc((size_t)BL * D * 4);
  float*  q       = (float*)alloc((size_t)BL * D * 4);
  float*  k       = (float*)alloc((size_t)BL * D * 4);
  float*  v       = (float*)alloc((size_t)BL * D * 4);
  float*  beta    = (float*)alloc((size_t)BL * H * 4);
  float*  dlt     = (float*)alloc((size_t)BL * D * 4);
  float*  f31     = (float*)alloc((size_t)BL * D * 4);
  __bf16* hgb     = (__bf16*)alloc((size_t)BL * 1024 * 2);
  float*  logits  = (float*)alloc((size_t)BL * 24 * 4);
  __bf16* fused   = (__bf16*)alloc((size_t)BL * D * 2);
  // fir buffers reuse the dead linear-projection buffers
  float* f1 = q_lin;
  float* f3 = k_lin;
  float* f7 = v_lin;

  // 1. conversions
  convert_hs_kernel<<<(BL * D) / 256, 256, 0, stream>>>(hs, hs_bf, gate_in, BL, D);
  convert_wT_kernel<<<(D * D) / 256, 256, 0, stream>>>(qw, wq_t, D, D);
  convert_wT_kernel<<<(D * D) / 256, 256, 0, stream>>>(kw, wk_t, D, D);
  convert_wT_kernel<<<(D * D) / 256, 256, 0, stream>>>(vw, wv_t, D, D);
  convert_wT_kernel<<<(D * D) / 256, 256, 0, stream>>>(ow, wo_t, D, D);
  convert_wT_kernel<<<(1120 * 1024) / 256, 256, 0, stream>>>(w1, w1_t, 1120, 1024);

  // 2. q/k/v projections (WMMA bf16, 128x128 blocks)
  dim3 gGemm(D / 128, BL / 128);  // (8, 32)
  gemm_bf16_kernel<0><<<gGemm, 256, 0, stream>>>(hs_bf, wq_t, q_lin, nullptr, nullptr, BL, D, D);
  gemm_bf16_kernel<0><<<gGemm, 256, 0, stream>>>(hs_bf, wk_t, k_lin, nullptr, nullptr, BL, D, D);
  gemm_bf16_kernel<0><<<gGemm, 256, 0, stream>>>(hs_bf, wv_t, v_lin, nullptr, nullptr, BL, D, D);

  // 3. causal conv + silu
  conv_silu_kernel<<<(BL * D) / 256, 256, 0, stream>>>(q_lin, qcw, q, B, L, D);
  conv_silu_kernel<<<(BL * D) / 256, 256, 0, stream>>>(k_lin, kcw, k, B, L, D);
  conv_silu_kernel<<<(BL * D) / 256, 256, 0, stream>>>(v_lin, vcw, v, B, L, D);

  // 4. beta
  beta_kernel<<<(BL * H) / 256, 256, 0, stream>>>(hs, bw, beta, BL, D, H);

  // 5. chunkwise delta rule (f32 WMMA on LDS state)
  size_t deltaLds = (size_t)(256 * 64 + 4 * 32 * 256 + 2 * 32 * 64 + 3 * 32 * 32 + 32) * 4;
  (void)hipFuncSetAttribute((const void*)delta_kernel,
                            hipFuncAttributeMaxDynamicSharedMemorySize, (int)deltaLds);
  delta_kernel<<<dim3(B * H, 4), 256, deltaLds, stream>>>(q, k, v, beta, dlt, B, L, H);

  // 6. FIR branches + branch stats -> gate_in tail
  fir_stats_kernel<<<BL * H, 256, 0, stream>>>(v, dlt, fw1, fw3, fw7, fw31,
                                               f1, f3, f7, f31, gate_in, B, L, H);

  // 7. gate MLP (bias+GELU fused into GEMM epilogue, bf16 out)
  gemm_bf16_kernel<1><<<gGemm, 256, 0, stream>>>(gate_in, w1_t, nullptr, hgb, b1, BL, 1024, 1120);
  logits_kernel<<<BL, 256, 0, stream>>>(hgb, w2, b2, logits, 1024, 24);

  // 8. softmax gate + fuse + RMSNorm
  fuse_kernel<<<BL * H, 256, 0, stream>>>(f1, f3, f7, f31, dlt, v, logits, glt, onw,
                                          fused, B, L, H);

  // 9. output projection
  gemm_bf16_kernel<0><<<gGemm, 256, 0, stream>>>(fused, wo_t, out, nullptr, nullptr, BL, D, D);
}
